// JastrowFactorGraph_40870908789024
// MI455X (gfx1250) — matrix operand order repaired and checked by reading, not compiled, not run on hardware
//
#include <hip/hip_runtime.h>
#include <hip/hip_bf16.h>
#include <math.h>

typedef _Float16 half16 __attribute__((ext_vector_type(16)));
typedef _Float16 half2v __attribute__((ext_vector_type(2)));
typedef float    float8 __attribute__((ext_vector_type(8)));

#define FEAT 64
#define NLAYERS 2

// Branch-free tanh: sign(x) * (1 - e)/(1 + e), e = exp(-2|x|) in (0,1].
// Denominator in (1,2] -> raw v_rcp_f32 is accurate; large |x| -> e=0 -> +/-1.
__device__ __forceinline__ float fast_tanhf(float x) {
  float ax = __builtin_fabsf(x);
  float e  = __expf(-2.0f * ax);
  float r  = (1.0f - e) * __builtin_amdgcn_rcpf(1.0f + e);
  return __builtin_copysignf(r, x);
}

// rbf center k of 64, cutoff 8.0
__device__ __forceinline__ float rbf_val(float d, int k) {
  float c = 8.0f * (float)k * (1.0f / 63.0f);
  float t = d - c;
  return __expf(-t * t);
}

// K index held by (vgpr j, half h) of a 16x32 f16 A fragment for this lane.
__device__ __forceinline__ int a_kidx(int j, int h, int lane) {
  int kh = (lane >> 4) * 8;
  return (j < 4) ? (kh + 2 * j + h) : (16 + kh + 2 * (j - 4) + h);
}

// B fragment (32x16 f16 slice, rows koff..koff+31, cols nt*16..+15) of a
// row-major 64x64 f32 matrix W.
__device__ __forceinline__ half16 load_bfrag(const float* __restrict__ W,
                                             int nt, int koff, int lane) {
  int ncol = nt * 16 + (lane & 15);
  int kbase = koff + (lane >> 4) * 8;
  half16 b = {};
#pragma unroll
  for (int j = 0; j < 8; ++j) {
#pragma unroll
    for (int h = 0; h < 2; ++h) {
      b[2 * j + h] = (_Float16)W[(kbase + 2 * j + h + (j >= 4 ? 8 : 0)) * FEAT + ncol];
    }
  }
  return b;
}

// A fragment pair (K=0..31 and K=32..63) from a row of f32 data in LDS.
// Per lane the K pattern is two contiguous runs of 8 -> 4x ds_load_b128 each.
__device__ __forceinline__ void load_afrag_f32row(const float* rowp, int lane,
                                                  half16& a0, half16& a1) {
  int kh = (lane >> 4) * 8;
  const float4 q0 = *(const float4*)(rowp + kh);
  const float4 q1 = *(const float4*)(rowp + kh + 4);
  const float4 q2 = *(const float4*)(rowp + 16 + kh);
  const float4 q3 = *(const float4*)(rowp + 16 + kh + 4);
  const float4 q4 = *(const float4*)(rowp + 32 + kh);
  const float4 q5 = *(const float4*)(rowp + 32 + kh + 4);
  const float4 q6 = *(const float4*)(rowp + 48 + kh);
  const float4 q7 = *(const float4*)(rowp + 48 + kh + 4);
  const float* p0 = (const float*)&q0;  // q0,q1 contiguous in struct order
  a0[0]=(_Float16)q0.x; a0[1]=(_Float16)q0.y; a0[2]=(_Float16)q0.z; a0[3]=(_Float16)q0.w;
  a0[4]=(_Float16)q1.x; a0[5]=(_Float16)q1.y; a0[6]=(_Float16)q1.z; a0[7]=(_Float16)q1.w;
  a0[8]=(_Float16)q2.x; a0[9]=(_Float16)q2.y; a0[10]=(_Float16)q2.z; a0[11]=(_Float16)q2.w;
  a0[12]=(_Float16)q3.x; a0[13]=(_Float16)q3.y; a0[14]=(_Float16)q3.z; a0[15]=(_Float16)q3.w;
  a1[0]=(_Float16)q4.x; a1[1]=(_Float16)q4.y; a1[2]=(_Float16)q4.z; a1[3]=(_Float16)q4.w;
  a1[4]=(_Float16)q5.x; a1[5]=(_Float16)q5.y; a1[6]=(_Float16)q5.z; a1[7]=(_Float16)q5.w;
  a1[8]=(_Float16)q6.x; a1[9]=(_Float16)q6.y; a1[10]=(_Float16)q6.z; a1[11]=(_Float16)q6.w;
  a1[12]=(_Float16)q7.x; a1[13]=(_Float16)q7.y; a1[14]=(_Float16)q7.z; a1[15]=(_Float16)q7.w;
  (void)p0;
}

// One workgroup = one batch element. Fully fused GNN for one graph.
// E edges (EPAD padded to 16), NN nodes (NPAD padded to 16), MAXDEG = max
// in-degree (EE: 29 exactly; EN: 30 for atoms, 10 for electrons).
template <int E, int EPAD, int NN, int NPAD, int MAXDEG>
__global__ void __launch_bounds__(256)
gnn_fused_kernel(const float* __restrict__ pos,    // [512,90]
                 const float* __restrict__ atoms,  // [10,3]
                 const int* __restrict__ src, const int* __restrict__ dst,
                 const int* __restrict__ types,
                 const float* __restrict__ emb,    // [T,64]
                 const float* __restrict__ wf,     // [64,64]
                 const float* __restrict__ bf,     // [64]
                 const float* __restrict__ wl,     // [2,64,64]
                 const float* __restrict__ bl,     // [2,64]
                 const float* __restrict__ wr,     // [64]
                 const float* __restrict__ br,     // [1]
                 float* __restrict__ outk)         // [512]
{
  __shared__ float    s_nodepos[NPAD * 3];
  __shared__ int      s_src[EPAD];
  __shared__ int      s_dst[EPAD];
  __shared__ float    s_d[EPAD];
  __shared__ _Float16 s_filt[EPAD * FEAT];
  __shared__ float    s_h[NPAD * FEAT];
  __shared__ float    s_agg[NPAD * FEAT];
  __shared__ int      s_inv[NN * MAXDEG];  // packed (src<<16)|edge
  __shared__ int      s_cnt[NN];           // in-degree counters
  __shared__ float    s_red;

  const int b    = blockIdx.x;
  const int tid  = threadIdx.x;
  const int lane = tid & 31;
  const int wave = tid >> 5;

  __builtin_prefetch(wf, 0, 1);   // global_prefetch_b8: weights into cache
  __builtin_prefetch(wl, 0, 1);

  // ---- phase 0: stage positions + edge lists; zero counters + agg pad ----
  for (int i = tid; i < NN * 3; i += 256) {
    s_nodepos[i] = (i < 90) ? pos[b * 90 + i] : atoms[i - 90];
  }
  for (int i = tid; i < EPAD; i += 256) {
    s_src[i] = (i < E) ? src[i] : 0;
    s_dst[i] = (i < E) ? dst[i] : 0;
  }
  for (int i = tid; i < NN; i += 256) s_cnt[i] = 0;
  for (int i = NN * FEAT + tid; i < NPAD * FEAT; i += 256) s_agg[i] = 0.0f;
  __syncthreads();

  // ---- phase 1: distances + embedding init + inverse adjacency build ----
  for (int e = tid; e < EPAD; e += 256) {
    float dd = 1.0e4f;  // pad rows -> rbf == 0 exactly
    if (e < E) {
      int s = s_src[e], t = s_dst[e];
      float dx = s_nodepos[3 * s + 0] - s_nodepos[3 * t + 0];
      float dy = s_nodepos[3 * s + 1] - s_nodepos[3 * t + 1];
      float dz = s_nodepos[3 * s + 2] - s_nodepos[3 * t + 2];
      dd = sqrtf(dx * dx + dy * dy + dz * dz);
    }
    s_d[e] = dd;
  }
  for (int i = tid; i < NPAD * FEAT; i += 256) {
    int node = i >> 6;
    s_h[i] = (node < NN) ? emb[types[node] * FEAT + (i & 63)] : 0.0f;
  }
  for (int e = tid; e < E; e += 256) {
    int dn = s_dst[e];
    int slot = atomicAdd(&s_cnt[dn], 1);      // ds_add_rtn_u32 (one-time)
    s_inv[dn * MAXDEG + slot] = e | (s_src[e] << 16);
  }
  __syncthreads();

  // ---- phase 2: filt = tanh(rbf @ wf + bf) via v_wmma_f32_16x16x32_f16 ----
  // Each wave OWNS an M-tile and sweeps all 4 N-tiles so the exp-built A
  // fragments are computed once and reused for 8 WMMAs.
  {
    half16 bw[4][2];
    float  bias[4];
#pragma unroll
    for (int nt = 0; nt < 4; ++nt) {
      bw[nt][0] = load_bfrag(wf, nt, 0, lane);
      bw[nt][1] = load_bfrag(wf, nt, 32, lane);
      bias[nt]  = bf[nt * 16 + (lane & 15)];
    }
    for (int mt = wave; mt < EPAD / 16; mt += 8) {
      float dv = s_d[mt * 16 + (lane & 15)];
      half16 a0 = {}, a1 = {};
#pragma unroll
      for (int j = 0; j < 8; ++j) {
#pragma unroll
        for (int h = 0; h < 2; ++h) {
          int k = a_kidx(j, h, lane);
          a0[2 * j + h] = (_Float16)rbf_val(dv, k);       // K = 0..31
          a1[2 * j + h] = (_Float16)rbf_val(dv, k + 32);  // K = 32..63
        }
      }
      int mbase = mt * 16 + (lane >> 4) * 8;
#pragma unroll
      for (int nt = 0; nt < 4; ++nt) {
        float8 c = {};
        c = __builtin_amdgcn_wmma_f32_16x16x32_f16(false, a0, false, bw[nt][0],
                                                   (short)0, c, false, false);
        c = __builtin_amdgcn_wmma_f32_16x16x32_f16(false, a1, false, bw[nt][1],
                                                   (short)0, c, false, false);
        int ncol = nt * 16 + (lane & 15);
#pragma unroll
        for (int r = 0; r < 8; ++r) {
          s_filt[(mbase + r) * FEAT + ncol] =
              (_Float16)fast_tanhf(c[r] + bias[nt]);
        }
      }
    }
  }
  __syncthreads();

  // ---- phase 3: message-passing layers (gather form, no atomics) ----
  for (int l = 0; l < NLAYERS; ++l) {
    // agg[n,f] = sum_{e : dst[e]==n} h[src[e],f] * filt[e,f]
    for (int idx = tid; idx < NN * 32; idx += 256) {
      int n = idx >> 5;
      int f = (idx & 31) * 2;
      int deg = s_cnt[n];
      const int* inv = &s_inv[n * MAXDEG];
      float acc0 = 0.0f, acc1 = 0.0f;
      for (int s = 0; s < deg; ++s) {
        int pk = inv[s];
        int e  = pk & 0xffff;
        int sn = pk >> 16;
        half2v fv = *(const half2v*)&s_filt[e * FEAT + f];
        const float2 hv = *(const float2*)&s_h[sn * FEAT + f];
        acc0 += hv.x * (float)fv[0];
        acc1 += hv.y * (float)fv[1];
      }
      *(float2*)&s_agg[n * FEAT + f] = make_float2(acc0, acc1);
    }
    __syncthreads();

    // h += tanh(agg @ wl[l] + bl[l])  via WMMA
    const float* Wl = wl + l * FEAT * FEAT;
    const float* Bl = bl + l * FEAT;
    for (int t = wave; t < (NPAD / 16) * 4; t += 8) {
      int mt = t >> 2, nt = t & 3;
      half16 b0 = load_bfrag(Wl, nt, 0, lane);
      half16 b1 = load_bfrag(Wl, nt, 32, lane);
      half16 a0, a1;
      load_afrag_f32row(&s_agg[(mt * 16 + (lane & 15)) * FEAT], lane, a0, a1);
      float8 c = {};
      c = __builtin_amdgcn_wmma_f32_16x16x32_f16(false, a0, false, b0,
                                                 (short)0, c, false, false);
      c = __builtin_amdgcn_wmma_f32_16x16x32_f16(false, a1, false, b1,
                                                 (short)0, c, false, false);
      int ncol  = nt * 16 + (lane & 15);
      float bias = Bl[ncol];
      int mbase = mt * 16 + (lane >> 4) * 8;
#pragma unroll
      for (int r = 0; r < 8; ++r) {
        s_h[(mbase + r) * FEAT + ncol] += fast_tanhf(c[r] + bias);
      }
    }
    __syncthreads();
  }

  // ---- phase 4: readout  sum_{n<NN,f} h[n,f]*wr[f] + br ----
  if (tid == 0) s_red = 0.0f;
  __syncthreads();
  float acc = 0.0f;
  for (int i = tid; i < NN * FEAT; i += 256) acc += s_h[i] * wr[i & 63];
  atomicAdd(&s_red, acc);
  __syncthreads();
  if (tid == 0) outk[b] = s_red + br[0];
}

__global__ void combine_kernel(const float* __restrict__ eek,
                               const float* __restrict__ enk,
                               float* __restrict__ out, int n) {
  int i = blockIdx.x * blockDim.x + threadIdx.x;
  if (i < n) out[i] = __expf(eek[i] + enk[i]);
}

extern "C" void kernel_launch(void* const* d_in, const int* in_sizes, int n_in,
                              void* d_out, int out_size, void* d_ws, size_t ws_size,
                              hipStream_t stream) {
  const float* pos    = (const float*)d_in[0];
  const float* atoms  = (const float*)d_in[1];
  const float* emb_ee = (const float*)d_in[2];
  const float* wf_ee  = (const float*)d_in[3];
  const float* bf_ee  = (const float*)d_in[4];
  const float* wl_ee  = (const float*)d_in[5];
  const float* bl_ee  = (const float*)d_in[6];
  const float* wr_ee  = (const float*)d_in[7];
  const float* br_ee  = (const float*)d_in[8];
  const float* emb_en = (const float*)d_in[9];
  const float* wf_en  = (const float*)d_in[10];
  const float* bf_en  = (const float*)d_in[11];
  const float* wl_en  = (const float*)d_in[12];
  const float* bl_en  = (const float*)d_in[13];
  const float* wr_en  = (const float*)d_in[14];
  const float* br_en  = (const float*)d_in[15];
  const int* ee_src   = (const int*)d_in[16];
  const int* ee_dst   = (const int*)d_in[17];
  const int* ee_types = (const int*)d_in[18];
  const int* en_src   = (const int*)d_in[19];
  const int* en_dst   = (const int*)d_in[20];
  const int* en_types = (const int*)d_in[21];

  float* eek = (float*)d_ws;       // [512]
  float* enk = eek + 512;          // [512]
  float* out = (float*)d_out;      // [512,1] f32

  // elec-elec: 870 edges (55 tiles), 30 nodes (pad 32), in-degree 29
  gnn_fused_kernel<870, 880, 30, 32, 29><<<512, 256, 0, stream>>>(
      pos, atoms, ee_src, ee_dst, ee_types,
      emb_ee, wf_ee, bf_ee, wl_ee, bl_ee, wr_ee, br_ee, eek);

  // elec-nuc: 600 edges (38 tiles), 40 nodes (pad 48), max in-degree 30
  gnn_fused_kernel<600, 608, 40, 48, 30><<<512, 256, 0, stream>>>(
      pos, atoms, en_src, en_dst, en_types,
      emb_en, wf_en, bf_en, wl_en, bl_en, wr_en, br_en, enk);

  combine_kernel<<<2, 256, 0, stream>>>(eek, enk, out, 512);
}
